// MultiGraphConvLayer_5669356834055
// MI455X (gfx1250) — matrix operand order repaired and checked by reading, not compiled, not run on hardware
//
#include <hip/hip_runtime.h>

// ---------------------------------------------------------------------------
// Types / helpers
// ---------------------------------------------------------------------------
typedef __attribute__((ext_vector_type(16))) __bf16          v16bf;
typedef __attribute__((ext_vector_type(16))) unsigned short  v16u;
typedef __attribute__((ext_vector_type(8)))  unsigned short  v8u;
typedef __attribute__((ext_vector_type(4)))  unsigned short  v4u;
typedef __attribute__((ext_vector_type(8)))  float           v8f;

#define DEV __device__ __forceinline__

// Scheduler pipeline hints (interleave DS/VMEM reads with WMMAs so the
// allocator doesn't hoist all fragment loads and spill accumulators).
#if __has_builtin(__builtin_amdgcn_sched_group_barrier)
#define SGB(mask, size) __builtin_amdgcn_sched_group_barrier((mask), (size), 0)
#else
#define SGB(mask, size)
#endif
// masks: 0x008 = MFMA/WMMA, 0x020 = VMEM read, 0x100 = DS read

DEV unsigned short f2bf(float f) {          // hardware cvt (RNE)
    __bf16 h = (__bf16)f;
    return __builtin_bit_cast(unsigned short, h);
}
DEV float bf2f(unsigned short u) {
    return (float)__builtin_bit_cast(__bf16, u);
}

DEV v16u cat16(v8u lo, v8u hi) {
    return __builtin_shufflevector(lo, hi, 0, 1, 2, 3, 4, 5, 6, 7,
                                           8, 9, 10, 11, 12, 13, 14, 15);
}
// A-fragment: row-major 16x32 tile row; elem e -> k = 16*half+e
DEV v16u loadA(const unsigned short* row, int half) {
    v8u lo = *(const v8u*)(row + 16 * half);
    v8u hi = *(const v8u*)(row + 16 * half + 8);
    return cat16(lo, hi);
}
// B-fragment: 16 contiguous k-values for one column; elem e -> k = 16*half+e
DEV v16u loadB(const unsigned short* p) {
    v8u lo = *(const v8u*)p;
    v8u hi = *(const v8u*)(p + 8);
    return cat16(lo, hi);
}
DEV v8f wmma_bf16(v16u a, v16u b, v8f c) {
    return __builtin_amdgcn_wmma_f32_16x16x32_bf16(
        false, __builtin_bit_cast(v16bf, a),
        false, __builtin_bit_cast(v16bf, b),
        (short)0, c, false, false);
}

// ---- CDNA5 async global->LDS copy (ASYNCcnt path, CDNA5 ISA) ------------
DEV void async_b128(const void* lds_ptr, const void* gptr) {
    unsigned l = (unsigned)(uintptr_t)lds_ptr;      // generic LDS addr [31:0] = LDS offset
    unsigned long long g = (unsigned long long)(uintptr_t)gptr;
    asm volatile("global_load_async_to_lds_b128 %0, %1, off"
                 :: "v"(l), "v"(g) : "memory");
}
DEV void wait_async0() {
#if __has_builtin(__builtin_amdgcn_s_wait_asynccnt)
    __builtin_amdgcn_s_wait_asynccnt(0);
#else
    asm volatile("s_wait_asynccnt 0x0" ::: "memory");
#endif
}

static constexpr int Bc = 4, Hc = 4, Nc = 2048, Dc = 128, HDc = 64;

// ---------------------------------------------------------------------------
// elementwise fp32 -> bf16
// ---------------------------------------------------------------------------
__global__ void __launch_bounds__(256) cvt_kernel(const float* __restrict__ s,
                                                  unsigned short* __restrict__ d, int n) {
    int i = blockIdx.x * 256 + threadIdx.x;
    if (i < n) d[i] = f2bf(s[i]);
}

// gcn [B,N,128] fp32 -> xb0T [B,128,N] bf16 (coalesced reads, scattered writes)
__global__ void __launch_bounds__(256) cvtT_kernel(const float* __restrict__ gcn,
                                                   unsigned short* __restrict__ xt) {
    int i = blockIdx.x * 256 + threadIdx.x;
    if (i >= Bc * Dc * Nc) return;
    int d = i & 127, n = (i >> 7) & (Nc - 1), b = i >> 18;
    xt[((size_t)b * Dc + d) * Nc + n] = f2bf(gcn[i]);
}

// xb1T[b,h,d,n]: d<128 -> xb0T[b,d,n] ; d>=128 -> bf16(g0T[b,h,d-128,n])
__global__ void __launch_bounds__(256) concat1_kernel(const unsigned short* __restrict__ xb0T,
                                                      const float* __restrict__ g0T,
                                                      unsigned short* __restrict__ xb1T) {
    size_t i = (size_t)blockIdx.x * 256 + threadIdx.x;
    if (i >= (size_t)Bc * Hc * 192 * Nc) return;
    int n = (int)(i & (Nc - 1));
    size_t rest = i >> 11;
    int d = (int)(rest % 192);
    size_t bh = rest / 192;
    int b = (int)(bh >> 2);
    xb1T[i] = (d < 128) ? xb0T[((size_t)b * Dc + d) * Nc + n]
                        : f2bf(g0T[(bh * 64 + (d - 128)) * Nc + n]);
}

// ---------------------------------------------------------------------------
// Fused GCN layer (per (b,h,128-row tile)):
//   Ax = adj_tile @ X       (bf16 WMMA; double-buffered LDS, async X staging,
//                            async bf16-adj staging in L1, reg-pipeline in L0)
//   L0: row-sums -> denom; adj bf16 -> L2-resident buffer; g -> g0T
//   S  = Ax + X[rows]; T = S @ w.T (WMMA); g = relu((T+2b)/denom)
//   both layers write bf16(g + gcn residual) straight into x2 concat matrix
// ---------------------------------------------------------------------------
template <int K, int L1, int ADJB>
__global__ void __launch_bounds__(256) gcn_layer_kernel(
    const float* __restrict__ adj, unsigned short* __restrict__ adjb,
    const unsigned short* __restrict__ xbT, const unsigned short* __restrict__ wb,
    const float* __restrict__ bias, const float* __restrict__ gcnf,
    float* __restrict__ den, float* __restrict__ goutT,
    unsigned short* __restrict__ x2b) {

    constexpr bool AADJ = (L1 && ADJB);     // adj staged via async (bf16 source)
    constexpr int KG  = K / 16;
    constexpr int KP  = K + 8;
    constexpr int XIA = K * 4 / 256;        // async X b128 segs per thread (2 or 3)

    __shared__ __align__(16) unsigned short sAdj[2][128][40];
    __shared__ __align__(16) unsigned short sXt[2][K][40];     // X transposed [n][k]
    __shared__ __align__(16) unsigned short sS[8][16][KP];
    __shared__ float sDen[128];

    const int tid  = threadIdx.x;
    const int wave = tid >> 5, lane = tid & 31;
    const int half = lane >> 4, c16 = lane & 15;
    const int b = blockIdx.z, h = blockIdx.y;
    const int rowBase = blockIdx.x * 128;

    const size_t aBase   = ((size_t)(b * Hc + h) * Nc + rowBase) * Nc;
    const size_t xTBase  = L1 ? (size_t)(b * Hc + h) * K * Nc : (size_t)b * K * Nc;
    const size_t denBase = (size_t)(b * Hc + h) * Nc;

    float4 af[4];
    float rs[4] = {0.f, 0.f, 0.f, 0.f};

    auto issueX = [&](int kk, int bf_) {
        #pragma unroll
        for (int q = 0; q < XIA; ++q) {
            int u = tid + 256 * q;
            int c = u >> 2, seg = u & 3;
            async_b128(&sXt[bf_][c][seg * 8],
                       xbT + xTBase + (size_t)c * Nc + kk + seg * 8);
        }
    };
    auto issueAdj = [&](int kk, int bf_) {
        #pragma unroll
        for (int q = 0; q < 2; ++q) {
            int u = tid + 256 * q;
            int r = u >> 2, c8 = (u & 3) * 8;
            async_b128(&sAdj[bf_][r][c8],
                       adjb + aBase + (size_t)r * Nc + kk + c8);
        }
    };
    auto loadAdjRegs = [&](int kk) {
        #pragma unroll
        for (int q = 0; q < 4; ++q) {
            int i4 = tid + 256 * q;
            int r = i4 >> 3, c4 = (i4 & 7) * 4;
            float4 v = *(const float4*)(adj + aBase + (size_t)r * Nc + kk + c4);
            af[q] = v;
            if (!L1) rs[q] += v.x + v.y + v.z + v.w;    // row r = tid>>3 + 32q
        }
        if (kk + 64 < Nc) {   // warm L2 two tiles ahead (global_prefetch)
            __builtin_prefetch(&adj[aBase + (size_t)(tid >> 1) * Nc + kk + 64 +
                                    (size_t)(tid & 1) * 16], 0, 1);
        }
    };
    auto storeAdj = [&](int kk, int bf_) {
        #pragma unroll
        for (int q = 0; q < 4; ++q) {
            int i4 = tid + 256 * q;
            int r = i4 >> 3, c4 = (i4 & 7) * 4;
            v4u s = {f2bf(af[q].x), f2bf(af[q].y), f2bf(af[q].z), f2bf(af[q].w)};
            *(v4u*)&sAdj[bf_][r][c4] = s;
            if (!L1 && ADJB)
                *(v4u*)(adjb + aBase + (size_t)r * Nc + kk + c4) = s;
        }
    };

    v8f acc[KG];
    #pragma unroll
    for (int g = 0; g < KG; ++g)
        #pragma unroll
        for (int i = 0; i < 8; ++i) acc[g][i] = 0.0f;

    // prologue: tile 0 into buffer 0
    if (AADJ) issueAdj(0, 0); else loadAdjRegs(0);
    issueX(0, 0);
    if (!AADJ) storeAdj(0, 0);
    wait_async0();
    __syncthreads();

    constexpr int T = Nc / 32;
    for (int t = 0; t < T; ++t) {
        const int cur = t & 1, nxt = cur ^ 1;
        const int kk2 = 32 * (t + 1);
        if (t + 1 < T) {                       // stage tile t+1 into other buffer
            if (AADJ) issueAdj(kk2, nxt); else loadAdjRegs(kk2);
            issueX(kk2, nxt);
        }

        const unsigned short* arow = &sAdj[cur][(wave << 4) + c16][0];
        v16u a = loadA(arow, half);
        #pragma unroll
        for (int g = 0; g < KG; ++g) {
            v16u bb = loadB(&sXt[cur][g * 16 + c16][16 * half]);
            acc[g] = wmma_bf16(a, bb, acc[g]);
        }
        // pipeline: A-frag reads, then per group {2 DS reads -> 1 WMMA}
        SGB(0x100, 2);
        #pragma unroll
        for (int g = 0; g < KG; ++g) { SGB(0x100, 2); SGB(0x008, 1); }

        if (t + 1 < T && !AADJ) storeAdj(kk2, nxt);
        wait_async0();
        __syncthreads();
    }

    // L0: finish row sums -> denom (threads with equal tid>>3 share rows)
    if (!L1) {
        #pragma unroll
        for (int q = 0; q < 4; ++q) {
            rs[q] += __shfl_xor(rs[q], 1, 32);
            rs[q] += __shfl_xor(rs[q], 2, 32);
            rs[q] += __shfl_xor(rs[q], 4, 32);
        }
        if ((lane & 7) == 0) {
            #pragma unroll
            for (int q = 0; q < 4; ++q) {
                int r = (tid >> 3) + 32 * q;
                float dv = rs[q] + 1.0f;
                sDen[r] = dv;
                den[denBase + rowBase + r] = dv;
            }
        }
        __syncthreads();
    }

    // S = Ax + X[row] -> per-wave LDS scratch (wave-private; DS in-order per wave)
    #pragma unroll
    for (int g = 0; g < KG; ++g) {
        #pragma unroll
        for (int r = 0; r < 8; ++r) {
            int m = r + (half << 3);
            int grow = rowBase + (wave << 4) + m;
            int d = g * 16 + c16;
            float xv = bf2f(xbT[xTBase + (size_t)d * Nc + grow]);
            sS[wave][m][d] = f2bf(acc[g][r] + xv);
        }
    }

    // T = S @ w.T   (16 x K) x (K x 64)
    v8f acc2[4];
    #pragma unroll
    for (int g = 0; g < 4; ++g)
        #pragma unroll
        for (int i = 0; i < 8; ++i) acc2[g][i] = 0.0f;

    #pragma unroll
    for (int k0 = 0; k0 < K; k0 += 32) {
        v16u a2 = loadA(&sS[wave][c16][k0], half);
        #pragma unroll
        for (int og = 0; og < 4; ++og) {
            v16u b2 = loadB(wb + (size_t)(h * HDc + og * 16 + c16) * K + k0 + 16 * half);
            acc2[og] = wmma_bf16(a2, b2, acc2[og]);
        }
    }
    #pragma unroll
    for (int kq = 0; kq < K / 32; ++kq) {
        SGB(0x100, 2);                                     // A-frag DS reads
        #pragma unroll
        for (int og = 0; og < 4; ++og) { SGB(0x020, 2); SGB(0x008, 1); }
    }

    float dn[8];
    #pragma unroll
    for (int r = 0; r < 8; ++r) {
        int lr = (wave << 4) + r + (half << 3);
        dn[r] = L1 ? den[denBase + rowBase + lr] : sDen[lr];
    }

    constexpr int OFF = L1 ? 64 : 0;
    #pragma unroll
    for (int og = 0; og < 4; ++og) {
        const int j  = og * 16 + c16;
        const float bj = 2.0f * bias[h * HDc + j];
        float vv[8];
        #pragma unroll
        for (int r = 0; r < 8; ++r) {
            float v = (acc2[og][r] + bj) / dn[r];
            vv[r] = v > 0.0f ? v : 0.0f;
        }
        // bf16(g + gcn residual) straight into head-major concat matrix
        #pragma unroll
        for (int r = 0; r < 8; ++r) {
            int grow = rowBase + (wave << 4) + r + (half << 3);
            int jg = OFF + j;
            x2b[((size_t)b * Nc + grow) * 512 + h * 128 + jg] =
                f2bf(vv[r] + gcnf[((size_t)b * Nc + grow) * 128 + jg]);
        }
        if (!L1) {   // g0 (transposed) for layer-1 input concat
            size_t gb = ((size_t)(b * Hc + h) * 64 + j) * Nc +
                        rowBase + (wave << 4) + (half << 3);
            float4 lo = {vv[0], vv[1], vv[2], vv[3]};
            float4 hi = {vv[4], vv[5], vv[6], vv[7]};
            *(float4*)(goutT + gb)     = lo;
            *(float4*)(goutT + gb + 4) = hi;
        }
    }
}

// ---------------------------------------------------------------------------
// Final: out[m, 0:128] = X2[m, 0:512] @ Wf.T + bf   (M = B*N = 8192)
// Fully async double-buffered staging (Wf's native [n][k] layout == sBt).
// ---------------------------------------------------------------------------
__global__ void __launch_bounds__(256) final_gemm_kernel(
    const unsigned short* __restrict__ x2, const unsigned short* __restrict__ wfb,
    const float* __restrict__ bf, float* __restrict__ out) {

    __shared__ __align__(16) unsigned short sA[2][128][40];
    __shared__ __align__(16) unsigned short sBt[2][128][40];

    const int tid  = threadIdx.x;
    const int wave = tid >> 5, lane = tid & 31;
    const int half = lane >> 4, c16 = lane & 15;
    const int mBase = blockIdx.x * 128;

    auto issue = [&](int kk, int bf_) {
        #pragma unroll
        for (int q = 0; q < 2; ++q) {
            int u = tid + 256 * q;
            int r = u >> 2, seg = u & 3;
            async_b128(&sA[bf_][r][seg * 8],
                       x2 + (size_t)(mBase + r) * 512 + kk + seg * 8);
        }
        #pragma unroll
        for (int q = 0; q < 2; ++q) {
            int u = tid + 256 * q;
            int n = u >> 2, seg = u & 3;
            async_b128(&sBt[bf_][n][seg * 8],
                       wfb + (size_t)n * 512 + kk + seg * 8);
        }
    };

    v8f acc[8];
    #pragma unroll
    for (int g = 0; g < 8; ++g)
        #pragma unroll
        for (int i = 0; i < 8; ++i) acc[g][i] = 0.0f;

    issue(0, 0);
    wait_async0();
    __syncthreads();

    for (int t = 0; t < 16; ++t) {
        const int cur = t & 1, nxt = cur ^ 1;
        if (t + 1 < 16) issue(32 * (t + 1), nxt);

        v16u a = loadA(&sA[cur][(wave << 4) + c16][0], half);
        #pragma unroll
        for (int g = 0; g < 8; ++g) {
            v16u bb = loadB(&sBt[cur][g * 16 + c16][16 * half]);
            acc[g] = wmma_bf16(a, bb, acc[g]);
        }
        SGB(0x100, 2);
        #pragma unroll
        for (int g = 0; g < 8; ++g) { SGB(0x100, 2); SGB(0x008, 1); }

        wait_async0();
        __syncthreads();
    }

    #pragma unroll
    for (int g = 0; g < 8; ++g) {
        #pragma unroll
        for (int r = 0; r < 8; ++r) {
            size_t m = (size_t)mBase + (wave << 4) + r + (half << 3);
            int n = g * 16 + c16;
            out[m * 128 + n] = acc[g][r] + bf[n];
        }
    }
}

// ---------------------------------------------------------------------------
// Launch
// ---------------------------------------------------------------------------
extern "C" void kernel_launch(void* const* d_in, const int* in_sizes, int n_in,
                              void* d_out, int out_size, void* d_ws, size_t ws_size,
                              hipStream_t stream) {
    (void)in_sizes; (void)n_in; (void)out_size;

    const float* adj = (const float*)d_in[0];   // [B,H,N,N]
    const float* gcn = (const float*)d_in[1];   // [B,N,128]
    const float* w0  = (const float*)d_in[2];   // [H,64,128]
    const float* b0  = (const float*)d_in[3];   // [H,64]
    const float* w1  = (const float*)d_in[4];   // [H,64,192]
    const float* b1  = (const float*)d_in[5];   // [H,64]
    const float* Wf  = (const float*)d_in[6];   // [128,512]
    const float* bf  = (const float*)d_in[7];   // [128]
    float* out = (float*)d_out;                 // [B,N,128]

    char* ws = (char*)d_ws;
    size_t o = 0;
    float*          den  = (float*)(ws + o);          o += (size_t)Bc*Hc*Nc*4;      // 128 KB
    unsigned short* xb0T = (unsigned short*)(ws + o); o += (size_t)Bc*Dc*Nc*2;      // 2 MB
    unsigned short* w0b  = (unsigned short*)(ws + o); o += (size_t)Hc*64*128*2;     // 64 KB
    unsigned short* w1b  = (unsigned short*)(ws + o); o += (size_t)Hc*64*192*2;     // 96 KB
    unsigned short* wfb  = (unsigned short*)(ws + o); o += (size_t)128*512*2;       // 128 KB
    float*          g0T  = (float*)(ws + o);          o += (size_t)Bc*Hc*64*Nc*4;   // 8 MB
    unsigned short* xb1T = (unsigned short*)(ws + o); o += (size_t)Bc*Hc*192*Nc*2;  // 12 MB
    unsigned short* x2b  = (unsigned short*)(ws + o); o += (size_t)Bc*Nc*512*2;     // 8 MB
    unsigned short* adjb = (unsigned short*)(ws + o);
    size_t need_adjb = o + (size_t)Bc*Hc*Nc*Nc*2;                                   // +134 MB
    const bool use_adjb = (ws_size >= need_adjb);   // constant per run -> deterministic

    // prep (bf16 operand copies; X transposed for async staging)
    cvtT_kernel<<<(Bc*Dc*Nc + 255)/256, 256, 0, stream>>>(gcn, xb0T);
    cvt_kernel<<<(Hc*64*128 + 255)/256, 256, 0, stream>>>(w0, w0b, Hc*64*128);
    cvt_kernel<<<(Hc*64*192 + 255)/256, 256, 0, stream>>>(w1, w1b, Hc*64*192);
    cvt_kernel<<<(128*512   + 255)/256, 256, 0, stream>>>(Wf, wfb, 128*512);

    dim3 lgrid(Nc / 128, Hc, Bc);
    if (use_adjb) {
        gcn_layer_kernel<128, 0, 1><<<lgrid, 256, 0, stream>>>(
            adj, adjb, xb0T, w0b, b0, gcn, den, g0T, x2b);
    } else {
        gcn_layer_kernel<128, 0, 0><<<lgrid, 256, 0, stream>>>(
            adj, adjb, xb0T, w0b, b0, gcn, den, g0T, x2b);
    }

    {
        size_t tot = (size_t)Bc*Hc*192*Nc;
        concat1_kernel<<<(unsigned)((tot + 255)/256), 256, 0, stream>>>(xb0T, g0T, xb1T);
    }

    if (use_adjb) {
        gcn_layer_kernel<192, 1, 1><<<lgrid, 256, 0, stream>>>(
            adj, adjb, xb1T, w1b, b1, gcn, den, nullptr, x2b);
    } else {
        gcn_layer_kernel<192, 1, 0><<<lgrid, 256, 0, stream>>>(
            adj, adjb, xb1T, w1b, b1, gcn, den, nullptr, x2b);
    }

    final_gemm_kernel<<<(Bc*Nc)/128, 256, 0, stream>>>(x2b, wfb, bf, out);
}
